// Bottleneck_75213467287669
// MI455X (gfx1250) — compile-verified, matrix operand order checked by reading, not run on hardware
//
#include <hip/hip_runtime.h>
#include <hip/hip_bf16.h>

// ---------------------------------------------------------------------------
// ShiftAddNet bottleneck for MI455X (gfx1250, wave32, WMMA).
//   shift convs  -> implicit-GEMM on v_wmma_f32_16x16x32_bf16 (pow-2 weights
//                   are exact in bf16; f32 accumulation). Wave tile 16x32
//                   (one A fragment feeds two WMMAs).
//   adder convs  -> LDS-tiled VALU  -sum|x - w|  (no matmul form exists)
// OOB / padded rows handled by pointer-swap to a zeroed scratch region so the
// WMMA loop has uniform control flow (no EXEC divergence in the hot loop).
// ---------------------------------------------------------------------------

typedef __attribute__((ext_vector_type(16))) __bf16      v16bf;
typedef __attribute__((ext_vector_type(8)))  float       v8f;
typedef __attribute__((ext_vector_type(8)))  unsigned    v8u;

#define NB   16     // batch
#define HH   28
#define WW_  28
#define HWN  784    // 28*28
#define C1   512    // input channels
#define PP   128    // planes
#define C3   512    // expansion*planes

__device__ __forceinline__ unsigned short f32_to_bf16u(float f) {
    unsigned u = __builtin_bit_cast(unsigned, f);
    u += 0x7FFFu + ((u >> 16) & 1u);          // round-to-nearest-even
    return (unsigned short)(u >> 16);
}

__device__ __forceinline__ float quant_pow2(float w) {
    if (w == 0.f) return 0.f;
    float a = fabsf(w) + 1e-8f;
    float r = rintf(log2f(a));                // jnp.round == RNE
    return copysignf(exp2f(r), w);
}

// ------------------------------- prep kernels ------------------------------

__global__ void k_zero16(unsigned short* __restrict__ z, int n) {
    int i = blockIdx.x * blockDim.x + threadIdx.x;
    if (i < n) z[i] = 0;
}

// pow-2 quantize, same layout [O][C] (1x1 convs), store bf16 bits
__global__ void k_quant(const float* __restrict__ w,
                        unsigned short* __restrict__ q, int n) {
    int i = blockIdx.x * blockDim.x + threadIdx.x;
    if (i < n) q[i] = f32_to_bf16u(quant_pow2(w[i]));
}

// w2s [O][C][3][3] -> bf16 [O][tap][C]   (tap = kh*3+kw)
__global__ void k_quant2(const float* __restrict__ w,
                         unsigned short* __restrict__ q) {
    int i = blockIdx.x * blockDim.x + threadIdx.x;
    const int n = PP * PP * 9;
    if (i >= n) return;
    int tap = i % 9, c = (i / 9) % PP, o = i / (9 * PP);
    q[(o * 9 + tap) * PP + c] = f32_to_bf16u(quant_pow2(w[i]));
}

// w2a [O][C][3][3] f32 -> f32 [O][tap][C]
__global__ void k_w2a_r(const float* __restrict__ w, float* __restrict__ q) {
    int i = blockIdx.x * blockDim.x + threadIdx.x;
    const int n = PP * PP * 9;
    if (i >= n) return;
    int tap = i % 9, c = (i / 9) % PP, o = i / (9 * PP);
    q[(o * 9 + tap) * PP + c] = w[i];
}

// fold BN: scale = g/sqrt(v+eps); bias = b - m*scale
__global__ void k_bnprep(const float* __restrict__ g, const float* __restrict__ b,
                         const float* __restrict__ m, const float* __restrict__ v,
                         float* __restrict__ s, float* __restrict__ bb, int n) {
    int i = blockIdx.x * blockDim.x + threadIdx.x;
    if (i >= n) return;
    float sc = g[i] / sqrtf(v[i] + 1e-5f);
    s[i] = sc;
    bb[i] = b[i] - m[i] * sc;
}

// x NCHW f32 -> NHWC bf16 bits
__global__ void k_xcvt(const float* __restrict__ x,
                       unsigned short* __restrict__ y) {
    int i = blockIdx.x * blockDim.x + threadIdx.x;
    const int n = NB * C1 * HWN;
    if (i >= n) return;
    int hw = i % HWN, c = (i / HWN) % C1, b = i / (HWN * C1);
    y[((size_t)b * HWN + hw) * C1 + c] = f32_to_bf16u(x[i]);
}

// ------------------------------ WMMA GEMM ---------------------------------
// Out[b,hw,o] = sum_{tap,c} A[b, hw+tap, c] * W[o, tap, c]    (A,W bf16)
// block = 256 thr = 8 waves; block tile 64(hw) x 64(o); wave tile 16x32:
// one A fragment drives two v_wmma (columns o0..15 and o0+16..31).
template<int C, int TAPS>
__global__ __launch_bounds__(256)
void k_gemm_bf16(const unsigned short* __restrict__ A,   // NHWC bf16 [B,784,C]
                 const unsigned short* __restrict__ Wt,  // bf16 [O, TAPS*C]
                 const unsigned short* __restrict__ zb,  // >= C*2+32 zero bytes
                 float* __restrict__ Out,                // NHWC f32 [B,784,O]
                 int O)
{
    constexpr int Kt = TAPS * C;
    const int tid  = threadIdx.x;
    const int lane = tid & 31;
    const int wid  = tid >> 5;
    const int hlf  = lane >> 4;      // half-wave selects K sub-chunk
    const int mrow = lane & 15;

    const int b   = blockIdx.z;
    const int hw0 = blockIdx.x * 64 + (wid & 3) * 16;
    const int o0  = blockIdx.y * 64 + (wid >> 2) * 32;

    const int hwA = hw0 + mrow;                 // A-fragment row (spatial)
    const int hA  = hwA / WW_;
    const int wA  = hwA % WW_;

    const unsigned short* Wr0 = Wt + (size_t)(o0 + mrow)      * Kt;
    const unsigned short* Wr1 = Wt + (size_t)(o0 + 16 + mrow) * Kt;

    v8f acc0 = {0.f, 0.f, 0.f, 0.f, 0.f, 0.f, 0.f, 0.f};
    v8f acc1 = {0.f, 0.f, 0.f, 0.f, 0.f, 0.f, 0.f, 0.f};

    #pragma unroll
    for (int t = 0; t < TAPS; ++t) {
        const int dh = (TAPS == 9) ? (t / 3 - 1) : 0;
        const int dw = (TAPS == 9) ? (t % 3 - 1) : 0;
        const int hh = hA + dh, ww = wA + dw;
        const bool valid = (hwA < HWN) & ((unsigned)hh < HH) & ((unsigned)ww < WW_);
        // pointer swap: padded / out-of-tile rows read zeros -> uniform flow
        const unsigned short* aptr =
            valid ? (A + ((size_t)b * HWN + (size_t)(hh * WW_ + ww)) * C) : zb;
        const unsigned short* b0p = Wr0 + t * C;
        const unsigned short* b1p = Wr1 + t * C;

        #pragma unroll 2
        for (int kk = 0; kk < C; kk += 32) {
            const int kb = kk + hlf * 8;        // ISA 16-bit layout:
            const uint4 a0  = *(const uint4*)(aptr + kb);        // K kb..kb+7
            const uint4 a1  = *(const uint4*)(aptr + kb + 16);   // K kb+16..+23
            const uint4 b00 = *(const uint4*)(b0p  + kb);
            const uint4 b01 = *(const uint4*)(b0p  + kb + 16);
            const uint4 b10 = *(const uint4*)(b1p  + kb);
            const uint4 b11 = *(const uint4*)(b1p  + kb + 16);
            if (kk + 32 < C)
                __builtin_prefetch(aptr + kb + 32, 0, 1);

            v8u au, bu0, bu1;
            au[0]  = a0.x;  au[1]  = a0.y;  au[2]  = a0.z;  au[3]  = a0.w;
            au[4]  = a1.x;  au[5]  = a1.y;  au[6]  = a1.z;  au[7]  = a1.w;
            bu0[0] = b00.x; bu0[1] = b00.y; bu0[2] = b00.z; bu0[3] = b00.w;
            bu0[4] = b01.x; bu0[5] = b01.y; bu0[6] = b01.z; bu0[7] = b01.w;
            bu1[0] = b10.x; bu1[1] = b10.y; bu1[2] = b10.z; bu1[3] = b10.w;
            bu1[4] = b11.x; bu1[5] = b11.y; bu1[6] = b11.z; bu1[7] = b11.w;
            const v16bf af  = __builtin_bit_cast(v16bf, au);
            const v16bf bf0 = __builtin_bit_cast(v16bf, bu0);
            const v16bf bf1 = __builtin_bit_cast(v16bf, bu1);

            acc0 = __builtin_amdgcn_wmma_f32_16x16x32_bf16(
                false, af, false, bf0, (short)0, acc0, false, false);
            acc1 = __builtin_amdgcn_wmma_f32_16x16x32_bf16(
                false, af, false, bf1, (short)0, acc1, false, false);
        }
    }

    // C/D layout: vgpr i -> row M = i + hlf*8, col N = mrow
    #pragma unroll
    for (int i = 0; i < 8; ++i) {
        const int hw = hw0 + i + hlf * 8;
        if (hw < HWN) {
            float* orow = Out + ((size_t)b * HWN + hw) * (size_t)O;
            orow[o0 + mrow]      = acc0[i];
            orow[o0 + 16 + mrow] = acc1[i];
        }
    }
}

// ------------------------------ adder convs -------------------------------
// out[b,hw,o] = -sum_{tap,c} |U[b, hw+tap, c] - W[o, tap, c]|
// block = 256 thr = 16(hw) x 16(o); LDS-staged 16x64 tiles of U and W.
#define TK 64

template<int C, int TAPS>
__device__ __forceinline__ float adder_acc(
    const float* __restrict__ U, const float* __restrict__ W,
    float (*su)[TK + 4], float (*sw)[TK + 4],
    int b, int hwT, int oT, int hw_l, int o_l)
{
    constexpr int Kt = TAPS * C;
    const int tid = threadIdx.x;
    float acc = 0.f;
    #pragma unroll
    for (int t = 0; t < TAPS; ++t) {
        const int dh = (TAPS == 9) ? (t / 3 - 1) : 0;
        const int dw = (TAPS == 9) ? (t % 3 - 1) : 0;
        for (int kk = 0; kk < C; kk += TK) {
            __syncthreads();
            #pragma unroll
            for (int e = tid; e < 16 * TK; e += 256) {
                const int r = e >> 6, c = e & (TK - 1);
                const int hw = hwT + r;
                const int hh = hw / WW_ + dh, ww = hw % WW_ + dw;
                float uv = 0.f;
                if ((unsigned)hh < HH && (unsigned)ww < WW_)
                    uv = U[((size_t)b * HWN + (size_t)(hh * WW_ + ww)) * C + kk + c];
                su[r][c] = uv;
                sw[r][c] = W[(size_t)(oT + r) * Kt + t * C + kk + c];
            }
            __syncthreads();
            #pragma unroll 4
            for (int k = 0; k < TK; k += 4) {
                const float4 u4 = *(const float4*)&su[hw_l][k];
                const float4 w4 = *(const float4*)&sw[o_l][k];
                acc += fabsf(u4.x - w4.x) + fabsf(u4.y - w4.y)
                     + fabsf(u4.z - w4.z) + fabsf(u4.w - w4.w);
            }
        }
    }
    return acc;
}

// adder + BN + ReLU, writes NHWC f32 AND bf16 (next GEMM input)
template<int C, int TAPS>
__global__ __launch_bounds__(256)
void k_adder_mid(const float* __restrict__ U, const float* __restrict__ W,
                 const float* __restrict__ sc, const float* __restrict__ bi,
                 float* __restrict__ OutF, unsigned short* __restrict__ OutB,
                 int O)
{
    __shared__ float su[16][TK + 4];
    __shared__ float sw[16][TK + 4];
    const int tid  = threadIdx.x;
    const int hw_l = tid >> 4, o_l = tid & 15;
    const int b = blockIdx.z, hwT = blockIdx.x * 16, oT = blockIdx.y * 16;

    float acc = adder_acc<C, TAPS>(U, W, su, sw, b, hwT, oT, hw_l, o_l);

    const int o = oT + o_l, hw = hwT + hw_l;
    float y = fmaf(-acc, sc[o], bi[o]);
    y = fmaxf(y, 0.f);
    const size_t idx = ((size_t)b * HWN + hw) * (size_t)O + o;
    OutF[idx] = y;
    OutB[idx] = f32_to_bf16u(y);
}

// adder + BN + identity residual (x, NCHW) + ReLU -> d_out NCHW f32
template<int C>
__global__ __launch_bounds__(256)
void k_adder_final(const float* __restrict__ U, const float* __restrict__ W,
                   const float* __restrict__ sc, const float* __restrict__ bi,
                   const float* __restrict__ X, float* __restrict__ Out,
                   int O)
{
    __shared__ float su[16][TK + 4];
    __shared__ float sw[16][TK + 4];
    const int tid  = threadIdx.x;
    const int hw_l = tid >> 4, o_l = tid & 15;
    const int b = blockIdx.z, hwT = blockIdx.x * 16, oT = blockIdx.y * 16;

    float acc = adder_acc<C, 1>(U, W, su, sw, b, hwT, oT, hw_l, o_l);

    const int o = oT + o_l, hw = hwT + hw_l;
    const size_t nchw = ((size_t)b * O + o) * HWN + hw;
    float y = fmaf(-acc, sc[o], bi[o]) + X[nchw];
    Out[nchw] = fmaxf(y, 0.f);
}

// ------------------------------- launcher ---------------------------------

extern "C" void kernel_launch(void* const* d_in, const int* in_sizes, int n_in,
                              void* d_out, int out_size, void* d_ws, size_t ws_size,
                              hipStream_t stream) {
    (void)in_sizes; (void)n_in; (void)out_size; (void)ws_size;
    const float* x   = (const float*)d_in[0];
    const float* w1s = (const float*)d_in[1];
    const float* w1a = (const float*)d_in[2];
    const float* w2s = (const float*)d_in[3];
    const float* w2a = (const float*)d_in[4];
    const float* w3s = (const float*)d_in[5];
    const float* w3a = (const float*)d_in[6];

    char* ws = (char*)d_ws;
    // workspace layout (bytes, all 256-aligned); ~57 MB total with reuse
    unsigned short* q1   = (unsigned short*)(ws + 0x0000000);  // 128K
    unsigned short* q2   = (unsigned short*)(ws + 0x0020000);  // 288K
    unsigned short* q3   = (unsigned short*)(ws + 0x0068000);  // 128K
    float*          w2ar = (float*)         (ws + 0x0088000);  // 576K
    float*          bnb  = (float*)         (ws + 0x0118000);  // 6K: s1,b1,s2,b2,s3,b3
    float *s1 = bnb, *b1 = bnb + 128, *s2 = bnb + 256, *b2 = bnb + 384,
          *s3 = bnb + 512, *b3 = bnb + 1024;
    unsigned short* zb   = (unsigned short*)(ws + 0x011A000);  // 8K zero region
    unsigned short* xb   = (unsigned short*)(ws + 0x0120000);  // 12.8M  x NHWC bf16
    float*          u12  = (float*)         (ws + 0x0D80000);  // 6.4M   u1 then u2
    float*          af   = (float*)         (ws + 0x1400000);  // 6.4M   a1/a2 f32
    unsigned short* ab   = (unsigned short*)(ws + 0x1A40000);  // 3.2M   a1/a2 bf16
    float*          u3   = (float*)         (ws + 0x1E00000);  // 25.7M

    const dim3 blk(256);
    // ---- prep ----
    k_zero16<<<dim3(16), blk, 0, stream>>>(zb, 4096);
    k_quant <<<dim3((PP * C1 + 255) / 256), blk, 0, stream>>>(w1s, q1, PP * C1);
    k_quant2<<<dim3((PP * PP * 9 + 255) / 256), blk, 0, stream>>>(w2s, q2);
    k_quant <<<dim3((C3 * PP + 255) / 256), blk, 0, stream>>>(w3s, q3, C3 * PP);
    k_w2a_r <<<dim3((PP * PP * 9 + 255) / 256), blk, 0, stream>>>(w2a, w2ar);
    k_bnprep<<<dim3(1), blk, 0, stream>>>((const float*)d_in[7],  (const float*)d_in[8],
                                          (const float*)d_in[9],  (const float*)d_in[10],
                                          s1, b1, PP);
    k_bnprep<<<dim3(1), blk, 0, stream>>>((const float*)d_in[11], (const float*)d_in[12],
                                          (const float*)d_in[13], (const float*)d_in[14],
                                          s2, b2, PP);
    k_bnprep<<<dim3(2), blk, 0, stream>>>((const float*)d_in[15], (const float*)d_in[16],
                                          (const float*)d_in[17], (const float*)d_in[18],
                                          s3, b3, C3);
    k_xcvt  <<<dim3((NB * C1 * HWN + 255) / 256), blk, 0, stream>>>(x, xb);

    // ---- conv1: 1x1 shift (512->128) then 1x1 adder (128->128) + bn1/relu
    k_gemm_bf16<C1, 1><<<dim3(13, PP / 64, NB), blk, 0, stream>>>(xb, q1, zb, u12, PP);
    k_adder_mid<PP, 1><<<dim3(HWN / 16, PP / 16, NB), blk, 0, stream>>>(
        u12, w1a, s1, b1, af, ab, PP);

    // ---- conv2: 3x3 shift then 3x3 adder + bn2/relu
    k_gemm_bf16<PP, 9><<<dim3(13, PP / 64, NB), blk, 0, stream>>>(ab, q2, zb, u12, PP);
    k_adder_mid<PP, 9><<<dim3(HWN / 16, PP / 16, NB), blk, 0, stream>>>(
        u12, w2ar, s2, b2, af, ab, PP);

    // ---- conv3: 1x1 shift (128->512) then 1x1 adder (512->512) + bn3 + res + relu
    k_gemm_bf16<PP, 1><<<dim3(13, C3 / 64, NB), blk, 0, stream>>>(ab, q3, zb, u3, C3);
    k_adder_final<C3><<<dim3(HWN / 16, C3 / 16, NB), blk, 0, stream>>>(
        u3, w3a, s3, b3, x, (float*)d_out, C3);
}
